// MiniBlock_2516850835849
// MI455X (gfx1250) — matrix operand order repaired and checked by reading, not compiled, not run on hardware
//
#include <hip/hip_runtime.h>

// ---------------------------------------------------------------------------
// MI455X (gfx1250) transformer block: LN -> MHA(causal) -> +res -> LN -> FFN -> +res
// B=4, T=2048, C=1024, H=16, D=64, F=4096.  bf16 WMMA everywhere, fp32 accum.
// ---------------------------------------------------------------------------

typedef __attribute__((ext_vector_type(16))) __bf16 v16bf;
typedef __attribute__((ext_vector_type(8)))  __bf16 v8bf;
typedef __attribute__((ext_vector_type(8)))  float  v8f;

static __device__ inline __bf16 f2bf(float f) {
  unsigned u = __builtin_bit_cast(unsigned, f);
  unsigned r = u + 0x7FFFu + ((u >> 16) & 1u);   // round-to-nearest-even
  unsigned short h = (unsigned short)(r >> 16);
  return __builtin_bit_cast(__bf16, h);
}

// ---- WMMA fragment loaders (ISA 7.12.2 layouts) ---------------------------
// A (16x32, 16-bit): lane&15 = row M; lanes 0-15 hold K {0..7,16..23},
// lanes 16-31 hold K {8..15,24..31}.  base points at row m0; ld in elements.
static __device__ inline v16bf load_frag_a(const __bf16* base, int ld, int k0) {
  int l = threadIdx.x & 31;
  const __bf16* p = base + (size_t)(l & 15) * ld + k0 + ((l >> 4) << 3);
  v8bf lo = *(const v8bf*)(p);        // K kb+0..7
  v8bf hi = *(const v8bf*)(p + 16);   // K kb+16..23
  v16bf f;
#pragma unroll
  for (int e = 0; e < 8; ++e) { f[e] = lo[e]; f[e + 8] = hi[e]; }
  return f;
}

// B (32x16, 16-bit): lane&15 = col N; lanes 0-15 hold K {0..7,16..23}, lanes 16-31 hold K {8..15,24..31}.
// baseT points at "row n0" of B^T (i.e. column n0 of B), ld = K-stride.
static __device__ inline v16bf load_frag_b(const __bf16* baseT, int ld, int k0) {
  int l = threadIdx.x & 31;
  const __bf16* p = baseT + (size_t)(l & 15) * ld + k0 + ((l >> 4) << 3);
  v8bf lo = *(const v8bf*)(p);
  v8bf hi = *(const v8bf*)(p + 16);
  v16bf f;
#pragma unroll
  for (int e = 0; e < 8; ++e) { f[e] = lo[e]; f[e + 8] = hi[e]; }
  return f;
}

static __device__ inline v8f wmma_bf16(v16bf a, v16bf b, v8f c) {
  return __builtin_amdgcn_wmma_f32_16x16x32_bf16(false, a, false, b,
                                                 (short)0, c, false, false);
}

// ---------------------------------------------------------------------------
// Weight convert + transpose: W[K][N] fp32 -> Wt[N][K] bf16
// ---------------------------------------------------------------------------
__global__ __launch_bounds__(256) void wt_kernel(const float* __restrict__ W,
                                                 __bf16* __restrict__ Wt,
                                                 int K, int N) {
  int idx = blockIdx.x * 256 + threadIdx.x;
  int k = idx / N, n = idx - k * N;            // coalesced read
  Wt[(size_t)n * K + k] = f2bf(W[idx]);
}

// ---------------------------------------------------------------------------
// LayerNorm: one 256-thread block per row of C=1024, bf16 output
// ---------------------------------------------------------------------------
__global__ __launch_bounds__(256) void ln_kernel(const float* __restrict__ x,
                                                 const float* __restrict__ g,
                                                 const float* __restrict__ bta,
                                                 __bf16* __restrict__ out, int C) {
  __shared__ float red[16];
  __shared__ float stats[2];
  int row = blockIdx.x, tid = threadIdx.x;
  int lane = tid & 31, wv = tid >> 5;
  const float4* xr = (const float4*)(x + (size_t)row * C);
  float4 v4 = xr[tid];
  float s  = v4.x + v4.y + v4.z + v4.w;
  float s2 = v4.x * v4.x + v4.y * v4.y + v4.z * v4.z + v4.w * v4.w;
#pragma unroll
  for (int off = 1; off < 32; off <<= 1) {
    s  += __shfl_xor(s, off, 32);
    s2 += __shfl_xor(s2, off, 32);
  }
  if (lane == 0) { red[wv] = s; red[8 + wv] = s2; }
  __syncthreads();
  if (tid == 0) {
    float ts = 0.f, ts2 = 0.f;
#pragma unroll
    for (int i = 0; i < 8; ++i) { ts += red[i]; ts2 += red[8 + i]; }
    float mu  = ts / (float)C;
    float var = ts2 / (float)C - mu * mu;
    stats[0] = mu;
    stats[1] = rsqrtf(var + 1e-5f);
  }
  __syncthreads();
  float mu = stats[0], rstd = stats[1];
  int c = tid * 4;
  float vv[4] = {v4.x, v4.y, v4.z, v4.w};
#pragma unroll
  for (int i = 0; i < 4; ++i)
    out[(size_t)row * C + c + i] = f2bf((vv[i] - mu) * rstd * g[c + i] + bta[c + i]);
}

// ---------------------------------------------------------------------------
// GEMM: out = A(bf16 MxK) @ B(bf16 KxN, supplied transposed NxK) [+bias][+res]
// mode 0: bf16 out; mode 1: relu -> bf16 out; mode 2: fp32 out = acc+bias+res
// 256 threads = 8 waves (2 m x 4 n); block tile 128x256; wave tile 64x64
// (4x4 wmma accums -> 16 v_wmma per 32-K step against 16 b128 loads).
// ---------------------------------------------------------------------------
__global__ __launch_bounds__(256) void gemm_bf16(
    const __bf16* __restrict__ A, const __bf16* __restrict__ Bt,
    const float* __restrict__ bias, const float* __restrict__ res,
    void* __restrict__ out, int M, int N, int K, int mode) {
  int tid = threadIdx.x;
  int wv = tid >> 5, lane = tid & 31;
  int wm = wv & 1, wn = wv >> 1;
  int m0 = blockIdx.y * 128 + wm * 64;
  int n0 = blockIdx.x * 256 + wn * 64;
  const __bf16* Abase = A  + (size_t)m0 * K;
  const __bf16* Bbase = Bt + (size_t)n0 * K;

  v8f acc[4][4];
#pragma unroll
  for (int i = 0; i < 4; ++i)
#pragma unroll
    for (int j = 0; j < 4; ++j)
#pragma unroll
      for (int e = 0; e < 8; ++e) acc[i][j][e] = 0.f;

  for (int k0 = 0; k0 < K; k0 += 32) {
    // speculative L2 prefetch of the next K-slab (safe past the end: dropped)
    __builtin_prefetch(Abase + (size_t)(lane & 15) * K + k0 + 32, 0, 3);
    __builtin_prefetch(Bbase + (size_t)(lane & 15) * K + k0 + 32, 0, 3);
    v16bf af[4], bfr[4];
#pragma unroll
    for (int i = 0; i < 4; ++i)
      af[i] = load_frag_a(Abase + (size_t)(i * 16) * K, K, k0);
#pragma unroll
    for (int j = 0; j < 4; ++j)
      bfr[j] = load_frag_b(Bbase + (size_t)(j * 16) * K, K, k0);
#pragma unroll
    for (int i = 0; i < 4; ++i)
#pragma unroll
      for (int j = 0; j < 4; ++j)
        acc[i][j] = wmma_bf16(af[i], bfr[j], acc[i][j]);
  }

  int nloc = lane & 15, mo = (lane >> 4) * 8;
#pragma unroll
  for (int i = 0; i < 4; ++i)
#pragma unroll
    for (int j = 0; j < 4; ++j)
#pragma unroll
      for (int v = 0; v < 8; ++v) {
        int mg = m0 + i * 16 + mo + v;
        int ng = n0 + j * 16 + nloc;
        float val = acc[i][j][v];
        if (bias) val += bias[ng];
        if (mode == 2) {
          ((float*)out)[(size_t)mg * N + ng] = val + res[(size_t)mg * N + ng];
        } else {
          if (mode == 1) val = fmaxf(val, 0.f);
          ((__bf16*)out)[(size_t)mg * N + ng] = f2bf(val);
        }
      }
}

// ---------------------------------------------------------------------------
// Flash-style causal attention.  Block = 128 thr (4 waves) per (b,h,64-query
// tile); wave w owns 16 query rows.  QK^T and PV on WMMA, online softmax fp32.
// ---------------------------------------------------------------------------
#define HS 64
__global__ __launch_bounds__(128) void attn_kernel(
    const __bf16* __restrict__ q, const __bf16* __restrict__ k,
    const __bf16* __restrict__ v, __bf16* __restrict__ o,
    int B, int T, int H) {
  __shared__ __bf16 Pl[4][16 * 64];   // per-wave P tile  (A-frag relayout)
  __shared__ __bf16 Vt[64 * 64];      // V tile transposed: [d][j]
  int b = blockIdx.z, h = blockIdx.y, qt = blockIdx.x;
  int tid = threadIdx.x, wv = tid >> 5, lane = tid & 31;
  int nloc = lane & 15, mo = (lane >> 4) * 8;
  int q0 = qt * 64;
  int C = H * HS;

  const __bf16* qb = q + (size_t)(b * T + q0 + wv * 16) * C + h * HS;
  v16bf qa0 = load_frag_a(qb, C, 0);
  v16bf qa1 = load_frag_a(qb, C, 32);

  v8f oacc[4];
#pragma unroll
  for (int dt = 0; dt < 4; ++dt)
#pragma unroll
    for (int e = 0; e < 8; ++e) oacc[dt][e] = 0.f;
  float rmax[8], rsum[8];
#pragma unroll
  for (int e = 0; e < 8; ++e) { rmax[e] = -INFINITY; rsum[e] = 0.f; }

  int qrow = q0 + wv * 16 + mo;       // + vv gives the global query row
  const float scale = 0.125f;         // 1/sqrt(64)
  int ntiles = qt + 1;                // causal: key tiles 0..qt

  for (int jt = 0; jt < ntiles; ++jt) {
    int j0 = jt * 64;
    __syncthreads();                  // previous PV reads done before restage
    for (int i = tid; i < 64 * 64; i += 128) {
      int j = i >> 6, d = i & 63;
      Vt[d * 64 + j] = v[(size_t)(b * T + j0 + j) * C + h * HS + d];
    }

    // ---- S = scale * Q K^T  (masked) ----
    float sv[4][8];
#pragma unroll
    for (int nt = 0; nt < 4; ++nt) {
      const __bf16* kb = k + (size_t)(b * T + j0 + nt * 16) * C + h * HS;
      v16bf b0 = load_frag_b(kb, C, 0);
      v16bf b1 = load_frag_b(kb, C, 32);
      v8f s;
#pragma unroll
      for (int e = 0; e < 8; ++e) s[e] = 0.f;
      s = wmma_bf16(qa0, b0, s);
      s = wmma_bf16(qa1, b1, s);
      int jg = j0 + nt * 16 + nloc;
#pragma unroll
      for (int vv = 0; vv < 8; ++vv) {
        float sc = s[vv] * scale;
        if (jg > qrow + vv) sc = -INFINITY;
        sv[nt][vv] = sc;
      }
    }

    // ---- online softmax (rows live across 16-lane halves) ----
#pragma unroll
    for (int vv = 0; vv < 8; ++vv) {
      float tmax = sv[0][vv];
#pragma unroll
      for (int nt = 1; nt < 4; ++nt) tmax = fmaxf(tmax, sv[nt][vv]);
#pragma unroll
      for (int off = 1; off < 16; off <<= 1)
        tmax = fmaxf(tmax, __shfl_xor(tmax, off, 32));
      float nm = fmaxf(rmax[vv], tmax);
      float corr = __expf(rmax[vv] - nm);
      rmax[vv] = nm;
      rsum[vv] *= corr;
#pragma unroll
      for (int dt = 0; dt < 4; ++dt) oacc[dt][vv] *= corr;
      float psum = 0.f;
#pragma unroll
      for (int nt = 0; nt < 4; ++nt) {
        float pp = __expf(sv[nt][vv] - nm);
        psum += pp;
        Pl[wv][(size_t)(mo + vv) * 64 + nt * 16 + nloc] = f2bf(pp);
      }
#pragma unroll
      for (int off = 1; off < 16; off <<= 1)
        psum += __shfl_xor(psum, off, 32);
      rsum[vv] += psum;
    }

    asm volatile("s_wait_dscnt 0" ::: "memory");  // own-wave P writes visible
    __syncthreads();                              // Vt staged by whole block

    // ---- O += P V ----
    v16bf pa0 = load_frag_a(&Pl[wv][0], 64, 0);
    v16bf pa1 = load_frag_a(&Pl[wv][0], 64, 32);
#pragma unroll
    for (int dt = 0; dt < 4; ++dt) {
      const __bf16* vb = &Vt[(size_t)(dt * 16) * 64];
      v16bf b0 = load_frag_b(vb, 64, 0);
      v16bf b1 = load_frag_b(vb, 64, 32);
      oacc[dt] = wmma_bf16(pa0, b0, oacc[dt]);
      oacc[dt] = wmma_bf16(pa1, b1, oacc[dt]);
    }
  }

  // ---- normalize + store bf16 ----
#pragma unroll
  for (int dt = 0; dt < 4; ++dt)
#pragma unroll
    for (int vv = 0; vv < 8; ++vv) {
      float val = oacc[dt][vv] / rsum[vv];
      size_t row = (size_t)(b * T + q0 + wv * 16 + mo + vv);
      o[row * C + h * HS + dt * 16 + nloc] = f2bf(val);
    }
}

// ---------------------------------------------------------------------------
extern "C" void kernel_launch(void* const* d_in, const int* in_sizes, int n_in,
                              void* d_out, int out_size, void* d_ws, size_t ws_size,
                              hipStream_t stream) {
  (void)in_sizes; (void)n_in; (void)out_size; (void)ws_size;
  const int Bb = 4, T = 2048, C = 1024, H = 16, F = 4096;
  const int M = Bb * T;

  const float* x     = (const float*)d_in[0];
  const float* Wq    = (const float*)d_in[1];
  const float* Wk    = (const float*)d_in[2];
  const float* Wv    = (const float*)d_in[3];
  const float* Wo    = (const float*)d_in[4];
  const float* bo    = (const float*)d_in[5];
  const float* ln1g  = (const float*)d_in[6];
  const float* ln1b  = (const float*)d_in[7];
  const float* ln2g  = (const float*)d_in[8];
  const float* ln2b  = (const float*)d_in[9];
  const float* W1    = (const float*)d_in[10];
  const float* b1    = (const float*)d_in[11];
  const float* W2    = (const float*)d_in[12];
  const float* b2    = (const float*)d_in[13];

  char* w = (char*)d_ws;
  auto alloc = [&](size_t bytes) {
    char* p = w;
    w += (bytes + 255) & ~(size_t)255;
    return p;
  };
  __bf16* xhat1 = (__bf16*)alloc((size_t)M * C * 2);
  __bf16* qb    = (__bf16*)alloc((size_t)M * C * 2);
  __bf16* kb    = (__bf16*)alloc((size_t)M * C * 2);
  __bf16* vb    = (__bf16*)alloc((size_t)M * C * 2);
  __bf16* attnb = (__bf16*)alloc((size_t)M * C * 2);
  float*  x1    = (float*) alloc((size_t)M * C * 4);
  __bf16* xhat2 = (__bf16*)alloc((size_t)M * C * 2);
  __bf16* h1    = (__bf16*)alloc((size_t)M * F * 2);
  __bf16* WqT   = (__bf16*)alloc((size_t)C * C * 2);
  __bf16* WkT   = (__bf16*)alloc((size_t)C * C * 2);
  __bf16* WvT   = (__bf16*)alloc((size_t)C * C * 2);
  __bf16* WoT   = (__bf16*)alloc((size_t)C * C * 2);
  __bf16* W1T   = (__bf16*)alloc((size_t)C * F * 2);
  __bf16* W2T   = (__bf16*)alloc((size_t)F * C * 2);

  // 1) weights -> bf16 transposed
  wt_kernel<<<(C * C) / 256, 256, 0, stream>>>(Wq, WqT, C, C);
  wt_kernel<<<(C * C) / 256, 256, 0, stream>>>(Wk, WkT, C, C);
  wt_kernel<<<(C * C) / 256, 256, 0, stream>>>(Wv, WvT, C, C);
  wt_kernel<<<(C * C) / 256, 256, 0, stream>>>(Wo, WoT, C, C);
  wt_kernel<<<(C * F) / 256, 256, 0, stream>>>(W1, W1T, C, F);
  wt_kernel<<<(F * C) / 256, 256, 0, stream>>>(W2, W2T, F, C);

  // 2) LN1
  ln_kernel<<<M, 256, 0, stream>>>(x, ln1g, ln1b, xhat1, C);

  // 3) QKV projections
  dim3 g1(C / 256, M / 128);
  gemm_bf16<<<g1, 256, 0, stream>>>(xhat1, WqT, nullptr, nullptr, qb, M, C, C, 0);
  gemm_bf16<<<g1, 256, 0, stream>>>(xhat1, WkT, nullptr, nullptr, kb, M, C, C, 0);
  gemm_bf16<<<g1, 256, 0, stream>>>(xhat1, WvT, nullptr, nullptr, vb, M, C, C, 0);

  // 4) causal attention
  dim3 ga(T / 64, H, Bb);
  attn_kernel<<<ga, 128, 0, stream>>>(qb, kb, vb, attnb, Bb, T, H);

  // 5) output projection + bias + residual -> x1 (fp32)
  gemm_bf16<<<g1, 256, 0, stream>>>(attnb, WoT, bo, x, x1, M, C, C, 2);

  // 6) LN2
  ln_kernel<<<M, 256, 0, stream>>>(x1, ln2g, ln2b, xhat2, C);

  // 7) FFN
  dim3 g2(F / 256, M / 128);
  gemm_bf16<<<g2, 256, 0, stream>>>(xhat2, W1T, b1, nullptr, h1, M, F, C, 1);
  gemm_bf16<<<g1, 256, 0, stream>>>(h1, W2T, b2, x1, (float*)d_out, M, C, F, 2);
}
